// RelationalGraphConvLayer_49804440765089
// MI455X (gfx1250) — matrix operand order repaired and checked by reading, not compiled, not run on hardware
//
#include <hip/hip_runtime.h>
#include <cstdint>

#define BB   8
#define RR   4
#define NN   1024
#define DIN  256
#define DOUT 256

typedef __attribute__((ext_vector_type(16))) _Float16 v16h;
typedef __attribute__((ext_vector_type(8)))  _Float16 v8h;
typedef __attribute__((ext_vector_type(4)))  _Float16 v4h;
typedef __attribute__((ext_vector_type(8)))  float    v8f;

// Matches the builtin's parameter pointee type exactly (gcc vector, 4 x i32).
typedef int v4i_vs __attribute__((vector_size(4 * sizeof(int))));

#if defined(__AMDGCN__) && defined(__has_builtin)
#if __has_builtin(__builtin_amdgcn_global_load_async_to_lds_b128)
#define HAVE_ASYNC_LDS 1
#endif
#endif
#ifndef HAVE_ASYNC_LDS
#define HAVE_ASYNC_LDS 0
#endif

#define AS1V(p) ((__attribute__((address_space(1))) v4i_vs*)(uintptr_t)(p))
#define AS3V(p) ((__attribute__((address_space(3))) v4i_vs*)(uintptr_t)(p))

__device__ __forceinline__ void wait_async_lds() {
#if defined(__AMDGCN__)
#if defined(__has_builtin) && __has_builtin(__builtin_amdgcn_s_wait_asynccnt)
    __builtin_amdgcn_s_wait_asynccnt(0);
#else
    asm volatile("s_wait_asynccnt 0x0" ::: "memory");
#endif
#endif
}

__device__ __forceinline__ void stage_b_tile(const _Float16* g, _Float16* l) {
#if HAVE_ASYNC_LDS
    __builtin_amdgcn_global_load_async_to_lds_b128(AS1V(g), AS3V(l), 0, 0);
    __builtin_amdgcn_global_load_async_to_lds_b128(AS1V(g), AS3V(l), 16, 0);
    __builtin_amdgcn_global_load_async_to_lds_b128(AS1V(g), AS3V(l), 32, 0);
    __builtin_amdgcn_global_load_async_to_lds_b128(AS1V(g), AS3V(l), 48, 0);
#else
    const uint4* src = (const uint4*)g;
    uint4* dst = (uint4*)l;
    dst[0] = src[0]; dst[1] = src[1]; dst[2] = src[2]; dst[3] = src[3];
#endif
}

__device__ __forceinline__ v16h ld_frag_lds(const _Float16* lo, const _Float16* hi) {
    v8h a = *(const v8h*)lo;   // 16B aligned ds_load_b128
    v8h b = *(const v8h*)hi;
    v16h r;
#pragma unroll
    for (int e = 0; e < 8; ++e) { r[e] = a[e]; r[8 + e] = b[e]; }
    return r;
}

// ---------------------------------------------------------------------------
// Kernel 1: r_inv[b,r,i] = 1 / sum_j adj[b,r,i,j]   (0 if the sum is 0)
// ---------------------------------------------------------------------------
__global__ void rowsum_inv_kernel(const float* __restrict__ adj,
                                  float* __restrict__ r_inv) {
    int row  = blockIdx.x * 8 + (threadIdx.x >> 5);
    int lane = threadIdx.x & 31;
    const float* p = adj + (size_t)row * NN;
    float s = 0.f;
#pragma unroll
    for (int t = 0; t < NN / 32; ++t) s += p[lane + 32 * t];
#pragma unroll
    for (int off = 16; off > 0; off >>= 1) s += __shfl_xor(s, off, 32);
    if (lane == 0) r_inv[row] = (s == 0.f) ? 0.f : 1.f / s;
}

// ---------------------------------------------------------------------------
// Kernel 2: Yt[b,r,n,j] = sum_d text[b,j,d] * weight[r,d,n]   (f16, transposed)
// ---------------------------------------------------------------------------
__global__ void ytmp_kernel(const float* __restrict__ text,
                            const float* __restrict__ weight,
                            _Float16* __restrict__ Yt) {
    int w    = blockIdx.x * 8 + (threadIdx.x >> 5);
    int lane = threadIdx.x & 31;
    const int tilesPerBR = (NN / 16) * (DOUT / 16);   // 1024
    int br = w / tilesPerBR;
    int t  = w % tilesPerBR;
    int b = br / RR, r = br % RR;
    int i0 = (t / (DOUT / 16)) * 16;
    int n0 = (t % (DOUT / 16)) * 16;
    int m = lane & 15, half = lane >> 4;
    int i = i0 + m;
    int n = n0 + (lane & 15);
    v8f c = {};
    for (int kt = 0; kt < DIN / 32; ++kt) {
        int k0 = kt * 32;
        const float* ap = text + ((size_t)b * NN + i) * DIN + k0 + half * 16;
        v16h a;
#pragma unroll
        for (int e = 0; e < 8; ++e) a[e]     = (_Float16)ap[e];
#pragma unroll
        for (int e = 0; e < 8; ++e) a[8 + e] = (_Float16)ap[8 + e];
        const float* bp = weight + ((size_t)r * DIN + k0 + half * 16) * DOUT + n;
        v16h bf;
#pragma unroll
        for (int e = 0; e < 16; ++e) bf[e] = (_Float16)bp[(size_t)e * DOUT];
        c = __builtin_amdgcn_wmma_f32_16x16x32_f16(false, a, false, bf,
                                                   (short)0, c, false, false);
    }
    _Float16* yp = Yt + ((size_t)(b * RR + r) * DOUT + n) * NN + i0 + half * 8;
#pragma unroll
    for (int v = 0; v < 8; ++v) yp[v] = (_Float16)c[v];
}

// ---------------------------------------------------------------------------
// Kernel 3: out[b,i,n] = sum_r sum_j (adj*r_inv) * Yt[b,r,n,j] + bias[n]
// 256-thread block = 8 waves (2 i-tiles x 4 n-groups) -> 32x256 tile.
// Double-buffered K-pipeline of 128 steps (K=32 each): async B copies and the
// A global load for step k+1 are in flight while step k's 4 WMMAs execute.
// ---------------------------------------------------------------------------
__global__ void __launch_bounds__(256)
rgcn_main_kernel(const float* __restrict__ adj,
                 const float* __restrict__ bias,
                 const float* __restrict__ r_inv,
                 const _Float16* __restrict__ Yt,
                 float* __restrict__ out) {
    // Row stride 40 halfs (80B): bank-conflict-free & 16B aligned.
    __shared__ _Float16 As[2][32 * 40];    //  2 x 2.5 KB
    __shared__ _Float16 Bs[2][256 * 40];   //  2 x 20  KB

    const int tid  = threadIdx.x;
    const int lane = tid & 31;
    const int wave = tid >> 5;
    const int iw   = wave >> 2;               // i-tile in block (0..1)
    const int ng   = wave & 3;                // 64-col group  (0..3)
    const int b    = blockIdx.x >> 5;
    const int I0   = (blockIdx.x & 31) * 32;

    const int m = lane & 15, half = lane >> 4;
    const int nl = ng * 64 + (lane & 15);

    // staging roles
    const int arow = tid >> 3;                // 0..31
    const int acol = (tid & 7) * 4;           // 0..28
    const int bn   = tid;                     // 0..255

    const int bRR = b * RR;
    // Per-r row scales for this staging thread's A row.
    const float s0 = r_inv[(size_t)(bRR + 0) * NN + I0 + arow];
    const float s1 = r_inv[(size_t)(bRR + 1) * NN + I0 + arow];
    const float s2 = r_inv[(size_t)(bRR + 2) * NN + I0 + arow];
    const float s3 = r_inv[(size_t)(bRR + 3) * NN + I0 + arow];

    const float* aBase0 =
        adj + (size_t)bRR * NN * NN + (size_t)(I0 + arow) * NN + acol;
    const _Float16* bBase0 = Yt + ((size_t)bRR * DOUT + bn) * NN;

    v8f c0 = {}, c1 = {}, c2 = {}, c3 = {};

    // ---- prologue: stage step 0 into buffer 0 ----
    {
        float4 av = *(const float4*)(aBase0);
        v4h aq;
        aq[0] = (_Float16)(av.x * s0);
        aq[1] = (_Float16)(av.y * s0);
        aq[2] = (_Float16)(av.z * s0);
        aq[3] = (_Float16)(av.w * s0);
        *(v4h*)(&As[0][arow * 40 + acol]) = aq;
        stage_b_tile(bBase0, &Bs[0][bn * 40]);
    }

    const int KSTEPS = RR * (NN / 32);   // 128
    for (int k = 0; k < KSTEPS; ++k) {
        wait_async_lds();     // this wave's async B(k) copies have landed
        __syncthreads();      // …and everyone's staging of buf[k&1] is visible

        const int cur = k & 1, nxt = cur ^ 1;
        const bool more = (k + 1 < KSTEPS);

        // ---- issue step k+1 loads (overlap with WMMAs below) ----
        float4 av; float sA = s0;
        if (more) {
            const int k1 = k + 1;
            const int r1 = k1 >> 5;
            const int jb1 = (k1 & 31) * 32;
            av = *(const float4*)(aBase0 + ((size_t)r1 << 20) + jb1);
            sA = (r1 < 2) ? (r1 == 0 ? s0 : s1) : (r1 == 2 ? s2 : s3);
            stage_b_tile(bBase0 + ((size_t)r1 << 18) + jb1, &Bs[nxt][bn * 40]);
            __builtin_prefetch(aBase0 + ((size_t)r1 << 20) + jb1 + 32, 0, 1);
        }

        // ---- consume buf[cur]: 1 A fragment + 4 B fragments -> 4 WMMAs ----
        const _Float16* ap = &As[cur][(iw * 16 + m) * 40];
        v16h a = ld_frag_lds(ap + half * 16, ap + half * 16 + 8);

        const _Float16* b0p = &Bs[cur][(size_t)nl * 40 + half * 16];
        v16h b0 = ld_frag_lds(b0p,           b0p + 8);
        v16h b1 = ld_frag_lds(b0p + 16 * 40, b0p + 16 * 40 + 8);
        v16h b2 = ld_frag_lds(b0p + 32 * 40, b0p + 32 * 40 + 8);
        v16h b3 = ld_frag_lds(b0p + 48 * 40, b0p + 48 * 40 + 8);

        c0 = __builtin_amdgcn_wmma_f32_16x16x32_f16(false, a, false, b0,
                                                    (short)0, c0, false, false);
        c1 = __builtin_amdgcn_wmma_f32_16x16x32_f16(false, a, false, b1,
                                                    (short)0, c1, false, false);
        c2 = __builtin_amdgcn_wmma_f32_16x16x32_f16(false, a, false, b2,
                                                    (short)0, c2, false, false);
        c3 = __builtin_amdgcn_wmma_f32_16x16x32_f16(false, a, false, b3,
                                                    (short)0, c3, false, false);

        // ---- finish A staging for step k+1 (waits the load issued above) ----
        if (more) {
            v4h aq;
            aq[0] = (_Float16)(av.x * sA);
            aq[1] = (_Float16)(av.y * sA);
            aq[2] = (_Float16)(av.z * sA);
            aq[3] = (_Float16)(av.w * sA);
            *(v4h*)(&As[nxt][arow * 40 + acol]) = aq;
        }
    }

    const float bv0 = bias[nl], bv1 = bias[nl + 16],
                bv2 = bias[nl + 32], bv3 = bias[nl + 48];
    float* op = out + ((size_t)b * NN + I0 + iw * 16) * DOUT;
#pragma unroll
    for (int v = 0; v < 8; ++v) {
        float* row = op + (size_t)(v + half * 8) * DOUT;   // C/D: VGPR v -> M=v(+8)
        row[nl]      = c0[v] + bv0;
        row[nl + 16] = c1[v] + bv1;
        row[nl + 32] = c2[v] + bv2;
        row[nl + 48] = c3[v] + bv3;
    }
}

// ---------------------------------------------------------------------------
extern "C" void kernel_launch(void* const* d_in, const int* in_sizes, int n_in,
                              void* d_out, int out_size, void* d_ws, size_t ws_size,
                              hipStream_t stream) {
    const float* text   = (const float*)d_in[0];   // [B,N,DIN]
    const float* adj    = (const float*)d_in[1];   // [B,R,N,N]
    const float* weight = (const float*)d_in[2];   // [R,DIN,DOUT]
    const float* bias   = (const float*)d_in[3];   // [DOUT]
    float* out = (float*)d_out;                    // [B,N,DOUT]

    // Workspace: r_inv (128 KB fp32) + Yt (16 MB f16, transposed)
    float*    r_inv = (float*)d_ws;
    _Float16* Yt    = (_Float16*)((char*)d_ws + (size_t)BB * RR * NN * sizeof(float));

    rowsum_inv_kernel<<<(BB * RR * NN) / 8, 256, 0, stream>>>(adj, r_inv);

    ytmp_kernel<<<(BB * RR * (NN / 16) * (DOUT / 16)) / 8, 256, 0, stream>>>(
        text, weight, Yt);

    rgcn_main_kernel<<<BB * (NN / 32), 256, 0, stream>>>(
        adj, bias, r_inv, Yt, out);
}
